// WindowAttention_2422361555566
// MI455X (gfx1250) — matrix operand order repaired and checked by reading, not compiled, not run on hardware
//
#include <hip/hip_runtime.h>
#include <math.h>

// ---------------- problem constants ----------------
#define EMBED_DIM 128
#define NUM_HEADS 4
#define WS        7
#define HEAD_DIM  32          // EMBED_DIM / NUM_HEADS
#define LQ        49          // WS*WS
#define NWIN      4096
#define QK_SCALE  0.17677669529663687f   // 32^-0.5

typedef unsigned short u16;
typedef __attribute__((ext_vector_type(16))) __bf16 v16bf;
typedef __attribute__((ext_vector_type(2)))  __bf16 v2bf;
typedef __attribute__((ext_vector_type(8)))  float  v8f;

union Frag16 { uint4 q[2]; v16bf v; };

// ---- f32 -> bf16 conversion -------------------------------------------------
// Prefer the hardware packed convert (1 VALU op / 2 elements); fall back to
// software round-to-nearest-even if the builtin is unavailable.
#if __has_builtin(__builtin_amdgcn_cvt_pk_bf16_f32)
static __device__ __forceinline__ unsigned bf16pk(float a, float b) {
    union { v2bf v; unsigned u; } u;
    u.v = __builtin_amdgcn_cvt_pk_bf16_f32(a, b);   // a -> low, b -> high
    return u.u;
}
static __device__ __forceinline__ u16 bf16c(float x) {
    return (u16)(bf16pk(x, 0.f) & 0xFFFFu);
}
#else
static __device__ __forceinline__ u16 bf16c(float x) {
    unsigned int u = __float_as_uint(x);
    u = u + 0x7FFFu + ((u >> 16) & 1u);
    return (u16)(u >> 16);
}
static __device__ __forceinline__ unsigned bf16pk(float a, float b) {
    return (unsigned)bf16c(a) | ((unsigned)bf16c(b) << 16);
}
#endif

// ---- WMMA fragment loaders (bf16, 16x16x32) -------------------------------
// A matrix 16x32, row-major source with pitch P (u16 units).
// lanes 0-15: row=lane,   elems 0-7 -> K=kb+0..7,  elems 8-15 -> K=kb+16..23
// lanes16-31: row=lane-16,elems 0-7 -> K=kb+8..15, elems 8-15 -> K=kb+24..31
static __device__ __forceinline__ v16bf load_a16(const u16* base, int pitch,
                                                 int row0, int kb, int lane) {
    const int hh = lane >> 4;
    const u16* rp = base + (row0 + (lane & 15)) * pitch;
    Frag16 f;
    f.q[0] = *(const uint4*)(rp + kb + hh * 8);
    f.q[1] = *(const uint4*)(rp + kb + 16 + hh * 8);
    return f.v;
}

// B matrix 32x16 from N-major storage ( [n][k] , pitch P in u16 units ).
// lanes 0-15: col=lane, K=kb+0..7,16..23 ; lanes 16-31: col=lane-16, K=kb+8..15,24..31
static __device__ __forceinline__ v16bf load_b16(const u16* base, int pitch,
                                                 int nb, int kb, int lane) {
    const int hh = lane >> 4;
    const u16* rp = base + (nb + (lane & 15)) * pitch + kb + hh * 8;
    Frag16 f;
    f.q[0] = *(const uint4*)(rp);
    f.q[1] = *(const uint4*)(rp + 16);
    return f.v;
}

static __device__ __forceinline__ v8f wmma_bf16(v16bf a, v16bf b, v8f c) {
    return __builtin_amdgcn_wmma_f32_16x16x32_bf16(false, a, false, b,
                                                   (short)0, c, false, false);
}

// ---------------- workspace layout (bytes) ----------------
#define WS_WQKVT   0                       // [384][128] bf16 N-major  (98304 B)
#define WS_WPROJT  98304                   // [128][128] bf16 N-major  (32768 B)
#define WS_BPERM   131072                  // [384] f32 permuted bias  ( 1536 B)
#define WS_RH      132608                  // [16][32] bf16 (13 used)  ( 1024 B)
#define WS_RW      133632                  // [16][32] bf16            ( 1024 B)

// ===========================================================================
// Kernel 0: convert + pre-swizzle weights into WMMA-friendly N-major bf16.
// ===========================================================================
__global__ void __launch_bounds__(256)
winattn_prep(const float* __restrict__ Wqkv, const float* __restrict__ bqkv,
             const float* __restrict__ Wproj,
             const float* __restrict__ rph, const float* __restrict__ rpw,
             u16* __restrict__ Wqkvt, u16* __restrict__ Wprojt,
             float* __restrict__ bperm, u16* __restrict__ Rh,
             u16* __restrict__ Rw) {
    const int stride = gridDim.x * blockDim.x;
    const int idx0   = blockIdx.x * blockDim.x + threadIdx.x;

    // W_qkv columns are (d,f,k) interleaved: c3 = d*12 + f*3 + k.
    // Re-lay as Wqkvt[n][c], n = k*128 + f*32 + d  (head-major q|k|v blocks).
    for (int i = idx0; i < 384 * 128; i += stride) {
        int n = i >> 7, c = i & 127;
        int qsel = n >> 7, within = n & 127;
        int f = within >> 5, d = within & 31;
        int c3 = d * 12 + f * 3 + qsel;
        Wqkvt[i] = bf16c(Wqkv[c * 384 + c3]);
    }
    for (int i = idx0; i < 384; i += stride) {
        int qsel = i >> 7, within = i & 127;
        int f = within >> 5, d = within & 31;
        bperm[i] = bqkv[d * 12 + f * 3 + qsel];
    }
    // W_proj transposed: Wprojt[n][c] = W_proj[c][n]
    for (int i = idx0; i < 128 * 128; i += stride) {
        int n = i >> 7, c = i & 127;
        Wprojt[i] = bf16c(Wproj[c * 128 + n]);
    }
    // rel-pos tables, padded 13 -> 16 rows (zeros)
    for (int i = idx0; i < 16 * 32; i += stride) {
        int r = i >> 5;
        Rh[i] = (r < 13) ? bf16c(rph[i]) : (u16)0;
        Rw[i] = (r < 13) ? bf16c(rpw[i]) : (u16)0;
    }
}

// ===========================================================================
// Kernel 1: one workgroup (8 wave32) per window.
// ===========================================================================
#define XP   136   // pitch (u16) for 64x128 row-major tiles (+8 pad: no bank conflicts)
#define VP   72    // pitch (u16) for transposed V / probs (64 + 8 pad)
#define SP   68    // pitch (f32) for scores

__global__ void __launch_bounds__(256)
winattn_main(const float* __restrict__ x,
             const u16* __restrict__ Wqkvt, const float* __restrict__ bperm,
             const u16* __restrict__ Wprojt, const float* __restrict__ bproj,
             const u16* __restrict__ Rh, const u16* __restrict__ Rw,
             float* __restrict__ out) {
    __shared__ u16   sX[64 * XP];               // x (bf16); reused as attn-out O
    __shared__ u16   sQ[64 * XP];
    __shared__ u16   sK[64 * XP];
    __shared__ u16   sVt[128 * VP];             // V transposed: [c][token]
    __shared__ float sS[NUM_HEADS * 64 * SP];   // raw scores
    __shared__ float sPh[2 * NUM_HEADS * 64 * 16]; // Q@Rh^T , Q@Rw^T
    __shared__ u16   sP[NUM_HEADS * 64 * VP];   // softmax probs (bf16)

    const int tid  = threadIdx.x;
    const int lane = tid & 31;
    const int wv   = tid >> 5;
    const int hh   = lane >> 4;
    const int blk  = blockIdx.x;
    const float* xg = x + (size_t)blk * (LQ * EMBED_DIM);

    // ---- Phase 0: load x tile (49x128), bf16, pad rows to 64 with zeros ----
    for (int i4 = tid; i4 < 64 * 32; i4 += 256) {
        int row = i4 >> 5, c4 = i4 & 31;
        float4 v = {0.f, 0.f, 0.f, 0.f};
        if (row < LQ) v = ((const float4*)xg)[row * 32 + c4];
        uint2 p;
        p.x = bf16pk(v.x, v.y);
        p.y = bf16pk(v.z, v.w);
        *(uint2*)(sX + row * XP + c4 * 4) = p;
    }
    __syncthreads();

    // ---- Phase 1: QKV = X @ Wqkv^T(+b).  M=64, N=384, K=128 ----------------
    for (int nt = wv; nt < 24; nt += 8) {
        v16bf bfr[4];
        #pragma unroll
        for (int kf = 0; kf < 4; ++kf)
            bfr[kf] = load_b16(Wqkvt, 128, nt * 16, kf * 32, lane);
        const int col  = nt * 16 + (lane & 15);
        const float bb = bperm[col];
        #pragma unroll
        for (int mt = 0; mt < 4; ++mt) {
            v8f c = {};
            #pragma unroll
            for (int kf = 0; kf < 4; ++kf)
                c = wmma_bf16(load_a16(sX, XP, mt * 16, kf * 32, lane), bfr[kf], c);
            if (col < 128) {                       // Q (row-major)
                #pragma unroll
                for (int e = 0; e < 8; ++e)
                    sQ[(mt * 16 + e + hh * 8) * XP + col] = bf16c(c[e] + bb);
            } else if (col < 256) {                // K (row-major)
                #pragma unroll
                for (int e = 0; e < 8; ++e)
                    sK[(mt * 16 + e + hh * 8) * XP + (col - 128)] = bf16c(c[e] + bb);
            } else {                               // V transposed, vector store
                uint4 p;
                p.x = bf16pk(c[0] + bb, c[1] + bb);
                p.y = bf16pk(c[2] + bb, c[3] + bb);
                p.z = bf16pk(c[4] + bb, c[5] + bb);
                p.w = bf16pk(c[6] + bb, c[7] + bb);
                *(uint4*)(sVt + (col - 256) * VP + mt * 16 + hh * 8) = p;
            }
        }
    }
    __syncthreads();

    // ---- Phase 2: S = Qh @ Kh^T, Ph = Qh @ relpos^T  (per head) ------------
    for (int j = wv; j < 96; j += 8) {
        if (j < 64) {                               // scores: 4 heads x 4x4 tiles
            int hd = j >> 4, rem = j & 15, mt = rem >> 2, ntk = rem & 3;
            v16bf a = load_a16(sQ, XP, mt * 16, hd * HEAD_DIM, lane);
            v16bf b = load_b16(sK, XP, ntk * 16, hd * HEAD_DIM, lane);
            v8f c = {};
            c = wmma_bf16(a, b, c);
            float* dst = sS + hd * 64 * SP;
            int colj = ntk * 16 + (lane & 15);
            #pragma unroll
            for (int e = 0; e < 8; ++e)
                dst[(mt * 16 + e + hh * 8) * SP + colj] = c[e];
        } else {                                    // rel-pos projections
            int idx = j - 64;
            int which = idx >> 4, rem = idx & 15, hd = rem >> 2, mt = rem & 3;
            const u16* R = which ? Rw : Rh;
            v16bf a = load_a16(sQ, XP, mt * 16, hd * HEAD_DIM, lane);
            v16bf b = load_b16(R, 32, 0, 0, lane);
            v8f c = {};
            c = wmma_bf16(a, b, c);
            float* dst = sPh + (which * NUM_HEADS + hd) * 64 * 16;
            int colj = (lane & 15);
            #pragma unroll
            for (int e = 0; e < 8; ++e)
                dst[(mt * 16 + e + hh * 8) * 16 + colj] = c[e];
        }
    }
    __syncthreads();

    // ---- Phase 3: softmax rows (196 rows = 4 heads x 49), f32 VALU ---------
    if (tid < NUM_HEADS * LQ) {
        const int hd = tid / LQ, i = tid % LQ;
        const int hi = i / WS, wi = i % WS;
        const float* Srow = sS + (hd * 64 + i) * SP;
        const float* PhR  = sPh + (0 * NUM_HEADS + hd) * 64 * 16 + i * 16;
        const float* PwR  = sPh + (1 * NUM_HEADS + hd) * 64 * 16 + i * 16;
        float m = -1e30f;
        for (int hj = 0; hj < WS; ++hj)
            for (int wj = 0; wj < WS; ++wj) {
                float v = Srow[hj * WS + wj] * QK_SCALE
                        + PhR[hi - hj + WS - 1] + PwR[wi - wj + WS - 1];
                m = fmaxf(m, v);
            }
        float s = 0.f;
        for (int hj = 0; hj < WS; ++hj)
            for (int wj = 0; wj < WS; ++wj) {
                float v = Srow[hj * WS + wj] * QK_SCALE
                        + PhR[hi - hj + WS - 1] + PwR[wi - wj + WS - 1];
                s += __expf(v - m);
            }
        const float inv = 1.f / s;
        u16* Prow = sP + (hd * 64 + i) * VP;
        for (int hj = 0; hj < WS; ++hj)
            for (int wj = 0; wj < WS; ++wj) {
                float v = Srow[hj * WS + wj] * QK_SCALE
                        + PhR[hi - hj + WS - 1] + PwR[wi - wj + WS - 1];
                Prow[hj * WS + wj] = bf16c(__expf(v - m) * inv);
            }
        for (int jj = LQ; jj < 64; ++jj) Prow[jj] = 0;   // pad key cols
    } else {
        int t2 = tid - NUM_HEADS * LQ;                    // zero pad query rows
        if (t2 < NUM_HEADS * (64 - LQ)) {
            int hd = t2 / (64 - LQ), row = LQ + t2 % (64 - LQ);
            u16* Prow = sP + (hd * 64 + row) * VP;
            for (int jj = 0; jj < 64; ++jj) Prow[jj] = 0;
        }
    }
    __syncthreads();

    // ---- Phase 4: O = P @ V  (per head: M=64, N=32, K=64) ------------------
    u16* sO = sX;                                   // reuse X region
    for (int j = wv; j < 32; j += 8) {
        int hd = j >> 3, rem = j & 7, mt = rem >> 1, nt2 = rem & 1;
        v8f c = {};
        #pragma unroll
        for (int kf = 0; kf < 2; ++kf) {
            v16bf a = load_a16(sP + hd * 64 * VP, VP, mt * 16, kf * 32, lane);
            v16bf b = load_b16(sVt, VP, hd * HEAD_DIM + nt2 * 16, kf * 32, lane);
            c = wmma_bf16(a, b, c);
        }
        int col = hd * HEAD_DIM + nt2 * 16 + (lane & 15);
        #pragma unroll
        for (int e = 0; e < 8; ++e)
            sO[(mt * 16 + e + hh * 8) * XP + col] = bf16c(c[e]);
    }
    __syncthreads();

    // ---- Phase 5: Y = O @ Wproj^T + b, write f32 to global -----------------
    float* og = out + (size_t)blk * (LQ * EMBED_DIM);
    for (int j = wv; j < 32; j += 8) {
        int mt = j >> 3, nt = j & 7;
        v8f c = {};
        #pragma unroll
        for (int kf = 0; kf < 4; ++kf) {
            v16bf a = load_a16(sO, XP, mt * 16, kf * 32, lane);
            v16bf b = load_b16(Wprojt, 128, nt * 16, kf * 32, lane);
            c = wmma_bf16(a, b, c);
        }
        int col = nt * 16 + (lane & 15);
        float bb = bproj[col];
        #pragma unroll
        for (int e = 0; e < 8; ++e) {
            int row = mt * 16 + e + hh * 8;
            if (row < LQ) og[row * EMBED_DIM + col] = c[e] + bb;
        }
    }
}

// ===========================================================================
extern "C" void kernel_launch(void* const* d_in, const int* in_sizes, int n_in,
                              void* d_out, int out_size, void* d_ws, size_t ws_size,
                              hipStream_t stream) {
    const float* x     = (const float*)d_in[0];
    const float* Wqkv  = (const float*)d_in[1];
    const float* bqkv  = (const float*)d_in[2];
    const float* Wproj = (const float*)d_in[3];
    const float* bproj = (const float*)d_in[4];
    const float* rph   = (const float*)d_in[5];
    const float* rpw   = (const float*)d_in[6];

    char* ws = (char*)d_ws;
    u16*   Wqkvt  = (u16*)(ws + WS_WQKVT);
    u16*   Wprojt = (u16*)(ws + WS_WPROJT);
    float* bperm  = (float*)(ws + WS_BPERM);
    u16*   Rh     = (u16*)(ws + WS_RH);
    u16*   Rw     = (u16*)(ws + WS_RW);

    winattn_prep<<<64, 256, 0, stream>>>(Wqkv, bqkv, Wproj, rph, rpw,
                                         Wqkvt, Wprojt, bperm, Rh, Rw);
    winattn_main<<<NWIN, 256, 0, stream>>>(x, Wqkvt, bperm, Wprojt, bproj,
                                           Rh, Rw, (float*)d_out);
}
